// MixMIMBlock_52398601011704
// MI455X (gfx1250) — matrix operand order repaired and checked by reading, not compiled, not run on hardware
//
#include <hip/hip_runtime.h>
#include <hip/hip_bf16.h>

// ---------------------------------------------------------------------------
// MixMIM / Swin block for gfx1250 (MI455X).
// bf16 WMMA (v_wmma_f32_16x16x32_bf16) for all GEMMs, f32 accumulate.
// GEMMs stage tiles through LDS with global_load_async_to_lds_b128 (ASYNCcnt).
// Attention stages bias + transposed-v tiles in LDS (vector ds traffic).
// ---------------------------------------------------------------------------

#define BATCH 16
#define HDIM 56
#define WDIM 56
#define CH 512
#define HEADS 16
#define WIN 7
#define NTOK 49          // tokens per window
#define NPAD 64          // padded tokens per window (WMMA tile multiple)
#define HID 2048
#define HD 32            // head dim
#define NWIN 1024        // BATCH * 8 * 8 windows
#define MWIN (NWIN * NPAD)         // 65536 padded window-rows
#define MTOK (BATCH * HDIM * WDIM) // 50176 tokens

typedef __bf16 v8bf  __attribute__((ext_vector_type(8)));
typedef __bf16 v16bf __attribute__((ext_vector_type(16)));
typedef float  v8f   __attribute__((ext_vector_type(8)));
typedef float  v4f   __attribute__((ext_vector_type(4)));

static __device__ __forceinline__ __bf16 f2bf(float f) {
    unsigned u = __builtin_bit_cast(unsigned, f);
    u += 0x7FFFu + ((u >> 16) & 1u);          // round-to-nearest-even
    unsigned short h = (unsigned short)(u >> 16);
    return __builtin_bit_cast(__bf16, h);
}

static __device__ __forceinline__ v16bf cat8(v8bf lo, v8bf hi) {
    return __builtin_shufflevector(lo, hi, 0,1,2,3,4,5,6,7,8,9,10,11,12,13,14,15);
}

static __device__ __forceinline__ float wave_sum(float v) {
#pragma unroll
    for (int o = 16; o > 0; o >>= 1) v += __shfl_xor(v, o, 32);
    return v;
}

// low 32 bits of a generic pointer to __shared__ = DS byte address
static __device__ __forceinline__ unsigned lds_lo(const void* p) {
    return (unsigned)(unsigned long long)p;
}

// per-lane 16B async copy global -> LDS (tracked by ASYNCcnt)
static __device__ __forceinline__ void async_b128(unsigned lds, const void* g) {
    asm volatile("global_load_async_to_lds_b128 %0, %1, off"
                 :: "v"(lds), "v"(g) : "memory");
}

// ---------------------------------------------------------------------------
// f32 -> bf16 elementwise (weight conversion)
// ---------------------------------------------------------------------------
__global__ void cvt_bf16_kernel(const float* __restrict__ in, __bf16* __restrict__ out, long n) {
    long i = (long)blockIdx.x * blockDim.x + threadIdx.x;
    long stride = (long)gridDim.x * blockDim.x;
    for (; i < n; i += stride) out[i] = f2bf(in[i]);
}

// ---------------------------------------------------------------------------
// Padded attention bias: bp[head][64][64]
//   col >= 49 -> -1e30 (masks pad keys), row >= 49 -> 0, else gathered bias.
// ---------------------------------------------------------------------------
__global__ __launch_bounds__(256) void bias_pad_kernel(
    const float* __restrict__ table, const int* __restrict__ idx, float* __restrict__ bp)
{
    const int i = blockIdx.x * 256 + threadIdx.x;      // 0 .. 16*4096-1
    const int head = i >> 12, r = (i >> 6) & 63, c = i & 63;
    float v;
    if (c >= NTOK)      v = -1e30f;
    else if (r >= NTOK) v = 0.f;
    else                v = table[idx[r * NTOK + c] * HEADS + head];
    bp[i] = v;
}

// ---------------------------------------------------------------------------
// LN1 + window partition + pad-to-64 + bf16 cast.  1 wave per padded row.
// ---------------------------------------------------------------------------
__global__ __launch_bounds__(256) void ln1_window_kernel(
    const float* __restrict__ x, const float* __restrict__ w, const float* __restrict__ b,
    __bf16* __restrict__ xw)
{
    const long g   = (long)blockIdx.x * 8 + (threadIdx.x >> 5); // padded window-row
    const int lane = threadIdx.x & 31;
    const long wi  = g >> 6;
    const int  p   = (int)(g & 63);
    __bf16* orow = xw + g * CH;
    if (p >= NTOK) {                      // zero pad rows
#pragma unroll
        for (int j = 0; j < 16; ++j) orow[lane + j * 32] = f2bf(0.f);
        return;
    }
    const int bb = (int)(wi >> 6), wrem = (int)(wi & 63);
    const int hh = (wrem >> 3) * WIN + p / WIN;
    const int ww = (wrem & 7) * WIN + p % WIN;
    const float* xr = x + ((long)bb * (HDIM * WDIM) + hh * WDIM + ww) * CH;

    float vbuf[16]; float s = 0.f;
#pragma unroll
    for (int j = 0; j < 16; ++j) { vbuf[j] = xr[lane + j * 32]; s += vbuf[j]; }
    const float mean = wave_sum(s) * (1.f / CH);
    float sq = 0.f;
#pragma unroll
    for (int j = 0; j < 16; ++j) { float d = vbuf[j] - mean; sq += d * d; }
    const float rstd = rsqrtf(wave_sum(sq) * (1.f / CH) + 1e-5f);
#pragma unroll
    for (int j = 0; j < 16; ++j) {
        int c = lane + j * 32;
        orow[c] = f2bf((vbuf[j] - mean) * rstd * w[c] + b[c]);
    }
}

// ---------------------------------------------------------------------------
// LN2 (token-major). 1 wave per row.
// ---------------------------------------------------------------------------
__global__ __launch_bounds__(256) void ln2_kernel(
    const float* __restrict__ x, const float* __restrict__ w, const float* __restrict__ b,
    __bf16* __restrict__ out)
{
    const long g = (long)blockIdx.x * 8 + (threadIdx.x >> 5);
    const int lane = threadIdx.x & 31;
    const float* xr = x + g * CH;
    __bf16* orow = out + g * CH;
    float vbuf[16]; float s = 0.f;
#pragma unroll
    for (int j = 0; j < 16; ++j) { vbuf[j] = xr[lane + j * 32]; s += vbuf[j]; }
    const float mean = wave_sum(s) * (1.f / CH);
    float sq = 0.f;
#pragma unroll
    for (int j = 0; j < 16; ++j) { float d = vbuf[j] - mean; sq += d * d; }
    const float rstd = rsqrtf(wave_sum(sq) * (1.f / CH) + 1e-5f);
#pragma unroll
    for (int j = 0; j < 16; ++j) {
        int c = lane + j * 32;
        orow[c] = f2bf((vbuf[j] - mean) * rstd * w[c] + b[c]);
    }
}

// ---------------------------------------------------------------------------
// bf16 WMMA GEMM with async-LDS double buffering.
//   C[m,n] = act( sum_k A[m,k]*Bw[n,k] + bias[n] (+resid) )
// Block = 256 thr = 8 waves. Tile: M=128 (wave = one 16-row strip), N=32
// (2 accumulators per wave, A fragment reused), K staged 64 at a time through
// LDS (row stride 72 bf16 -> conflict-free ds_load_b128 fragments).
// ---------------------------------------------------------------------------
#define AST 72   // LDS row stride in bf16 elements

template <bool GELU_, bool OUTBF, bool RESID>
__global__ __launch_bounds__(256) void gemm_bf16_kernel(
    const __bf16* __restrict__ A, const __bf16* __restrict__ Bw,
    const float* __restrict__ bias, void* __restrict__ Cout,
    const float* __restrict__ resid, int M, int Nn, int K)
{
    __shared__ __bf16 Ab[2][128 * AST];
    __shared__ __bf16 Bb[2][32 * AST];

    const int tid  = threadIdx.x;
    const int lane = tid & 31;
    const int wave = tid >> 5;                 // 0..7 = M strip
    const long mbase = (long)blockIdx.y * 128;
    const long nbase = (long)blockIdx.x * 32;
    const int half = lane >> 4, l15 = lane & 15;

    // ---- async stage of one K-slab (A:128x64, B:32x64) into buffer `buf` ----
    auto issue = [&](int buf, int k0) {
#pragma unroll
        for (int j = 0; j < 4; ++j) {          // A: 1024 16B-chunks / 256 thr
            int chunk = tid + j * 256;
            int row = chunk >> 2, col = (chunk & 3) * 16;
            async_b128(lds_lo(&Ab[buf][row * AST + col]),
                       A + (mbase + row) * (long)K + k0 + col);
        }
        {                                       // B: 256 8-elem chunks / 256 thr
            int row = tid >> 3, col = (tid & 7) * 8;
            async_b128(lds_lo(&Bb[buf][row * AST + col]),
                       Bw + (nbase + row) * (long)K + k0 + col);
        }
    };

    v8f acc0 = {}, acc1 = {};
    issue(0, 0);
    const int nk = K >> 6;                      // K / 64
    for (int kt = 0; kt < nk; ++kt) {
        if (kt + 1 < nk) {
            issue((kt + 1) & 1, (kt + 1) * 64);
            asm volatile("s_wait_asynccnt 5" ::: "memory");   // slab kt landed
        } else {
            asm volatile("s_wait_asynccnt 0" ::: "memory");
        }
        __syncthreads();
        const __bf16* Abuf = Ab[kt & 1];
        const __bf16* Bbuf = Bb[kt & 1];
        const __bf16* ar = Abuf + (wave * 16 + l15) * AST;
#pragma unroll
        for (int ks = 0; ks < 64; ks += 32) {
            v16bf a = cat8(*(const v8bf*)(ar + ks + half * 16),
                           *(const v8bf*)(ar + ks + half * 16 + 8));
            v16bf b0 = *(const v16bf*)(Bbuf + l15 * AST + ks + half * 16);
            v16bf b1 = *(const v16bf*)(Bbuf + (16 + l15) * AST + ks + half * 16);
            acc0 = __builtin_amdgcn_wmma_f32_16x16x32_bf16(false, a, false, b0,
                                                           (short)0, acc0, false, false);
            acc1 = __builtin_amdgcn_wmma_f32_16x16x32_bf16(false, a, false, b1,
                                                           (short)0, acc1, false, false);
        }
        __syncthreads();                        // buffer reuse fence
    }

#pragma unroll
    for (int ni = 0; ni < 2; ++ni) {
        const v8f acc = ni ? acc1 : acc0;
        const long col = nbase + ni * 16 + l15;
        const float bs = bias[col];
#pragma unroll
        for (int i = 0; i < 8; ++i) {
            long row = mbase + wave * 16 + half * 8 + i;
            if (row >= M) continue;
            float v = acc[i] + bs;
            if (GELU_) v = 0.5f * v * (1.f + erff(v * 0.70710678118654752f));
            if (RESID) v += resid[row * (long)Nn + col];
            if (OUTBF) ((__bf16*)Cout)[row * (long)Nn + col] = f2bf(v);
            else       ((float*)Cout)[row * (long)Nn + col] = v;
        }
    }
}

// ---------------------------------------------------------------------------
// Windowed attention: one block (128 thr = 4 waves) per (window, head).
// S = (q k^T)*scale + bias_pad -> softmax (LDS) -> O = P v
// qkv layout: (MWIN, 1536) bf16, cols = which*512 + head*32 + d.
// Bias tile and transposed v tile are staged in LDS at entry so the hot
// epilogues do only DS traffic.
// ---------------------------------------------------------------------------
#define VST 72   // LDS row stride (bf16) of transposed-v tile

__global__ __launch_bounds__(128) void attn_kernel(
    const __bf16* __restrict__ qkv, const float* __restrict__ bp,
    __bf16* __restrict__ attn_o)
{
    __shared__ float  S[64 * 65];      // scores / probabilities
    __shared__ float  Bsh[64 * 64];    // padded bias tile for this head
    __shared__ __bf16 Vsh[HD * VST];   // v^T: Vsh[d][k]

    const int wi = blockIdx.x >> 4;
    const int head = blockIdx.x & 15;
    const int tid  = threadIdx.x;
    const int lane = tid & 31;
    const int wave = tid >> 5;          // mi = wave (rows 16*wave..)
    const int half = lane >> 4, l15 = lane & 15;

    const __bf16* base = qkv + (long)wi * NPAD * (3 * CH) + head * HD;
    const __bf16* qp = base;
    const __bf16* kp = base + CH;
    const __bf16* vp = base + 2 * CH;
    const float* bph = bp + (head << 12);          // [64][64]
    const int m0 = wave * 16;

    // ---- stage 0: cooperative LDS staging ----
    // bias: 4096 floats, vectorized 16B loads (8 per thread)
#pragma unroll
    for (int j = 0; j < 8; ++j) {
        int e4 = tid + j * 128;                     // float4 index 0..1023
        *(v4f*)&Bsh[e4 * 4] = *(const v4f*)&bph[e4 * 4];
    }
    // v^T: read v (64 x 32 bf16) with 16B vector loads, scatter transposed
    {
        const int k = tid >> 1, d0 = (tid & 1) * 16;
        v8bf va = *(const v8bf*)(vp + (long)k * (3 * CH) + d0);
        v8bf vb = *(const v8bf*)(vp + (long)k * (3 * CH) + d0 + 8);
#pragma unroll
        for (int e = 0; e < 8; ++e) {
            Vsh[(d0 + e) * VST + k]     = va[e];
            Vsh[(d0 + 8 + e) * VST + k] = vb[e];
        }
    }

    // ---- stage 1: S tiles (K = HD = 32, single WMMA step each) ----
    const __bf16* qrow = qp + (long)(m0 + l15) * (3 * CH);
    v16bf aq = cat8(*(const v8bf*)(qrow + half * 16),
                    *(const v8bf*)(qrow + half * 16 + 8));
    const float scale = 0.17677669529663689f;      // 1/sqrt(32)

    __syncthreads();                                // Bsh/Vsh ready

#pragma unroll
    for (int nt = 0; nt < 4; ++nt) {
        const int n0 = nt * 16;
        v16bf bk = *(const v16bf*)(kp + (long)(n0 + l15) * (3 * CH) + half * 16);
        v8f c = {};
        c = __builtin_amdgcn_wmma_f32_16x16x32_bf16(false, aq, false, bk,
                                                    (short)0, c, false, false);
        const int kcol = n0 + l15;
#pragma unroll
        for (int i = 0; i < 8; ++i) {
            const int qr = m0 + half * 8 + i;
            S[qr * 65 + kcol] = c[i] * scale + Bsh[qr * 64 + kcol];
        }
    }
    __syncthreads();

    // ---- stage 2: row softmax in LDS ----
    if (tid < 64) {
        float* row = &S[tid * 65];
        if (tid < NTOK) {
            float mx = -1e30f;
            for (int j = 0; j < NTOK; ++j) mx = fmaxf(mx, row[j]);
            float s = 0.f;
            for (int j = 0; j < NTOK; ++j) { float e = __expf(row[j] - mx); row[j] = e; s += e; }
            float inv = 1.f / s;
            for (int j = 0; j < NTOK; ++j) row[j] *= inv;
            for (int j = NTOK; j < 64; ++j) row[j] = 0.f;
        } else {
            for (int j = 0; j < 64; ++j) row[j] = 0.f;
        }
    }
    __syncthreads();

    // ---- stage 3: O = P @ v  (K=64 tokens -> 2 WMMA steps; N=32 -> 2 tiles) ----
    const float* prow = &S[(m0 + l15) * 65];
#pragma unroll
    for (int nt = 0; nt < 2; ++nt) {
        const int dcol = nt * 16 + l15;
        const __bf16* vrow = &Vsh[dcol * VST];
        v8f c = {};
#pragma unroll
        for (int kk = 0; kk < 2; ++kk) {
            const int k0 = kk * 32;
            v16bf a2;
#pragma unroll
            for (int e = 0; e < 8; ++e) {
                a2[e]     = f2bf(prow[k0 + half * 16 + e]);
                a2[e + 8] = f2bf(prow[k0 + half * 16 + 8 + e]);
            }
            v16bf b2 = cat8(*(const v8bf*)(vrow + k0 + half * 16),
                            *(const v8bf*)(vrow + k0 + half * 16 + 8));
            c = __builtin_amdgcn_wmma_f32_16x16x32_bf16(false, a2, false, b2,
                                                        (short)0, c, false, false);
        }
#pragma unroll
        for (int i = 0; i < 8; ++i) {
            const int row = m0 + half * 8 + i;
            if (row < NTOK)
                attn_o[((long)wi * NPAD + row) * CH + head * HD + nt * 16 + l15] = f2bf(c[i]);
        }
    }
}

// ---------------------------------------------------------------------------
// Window reverse + residual: x_attn[token] = x[token] + proj[window-row]
// ---------------------------------------------------------------------------
__global__ __launch_bounds__(256) void revres_kernel(
    const float* __restrict__ x, const float* __restrict__ proj, float* __restrict__ x_attn)
{
    const long g = (long)blockIdx.x * 8 + (threadIdx.x >> 5);   // token 0..50175
    const int lane = threadIdx.x & 31;
    const int bb = (int)(g / (HDIM * WDIM));
    const int r  = (int)(g % (HDIM * WDIM));
    const int hh = r / WDIM, ww = r % WDIM;
    const long wrow = ((long)bb * 64 + (hh / WIN) * 8 + (ww / WIN)) * NPAD
                    + (hh % WIN) * WIN + (ww % WIN);
    const float* xr = x + g * CH;
    const float* pr = proj + wrow * CH;
    float* orow = x_attn + g * CH;
#pragma unroll
    for (int j = 0; j < 16; ++j) {
        int c = lane + j * 32;
        orow[c] = xr[c] + pr[c];
    }
}

// ---------------------------------------------------------------------------
// Launch
// ---------------------------------------------------------------------------
extern "C" void kernel_launch(void* const* d_in, const int* in_sizes, int n_in,
                              void* d_out, int out_size, void* d_ws, size_t ws_size,
                              hipStream_t stream) {
    const float* x        = (const float*)d_in[0];
    const float* norm1_w  = (const float*)d_in[1];
    const float* norm1_b  = (const float*)d_in[2];
    const float* qkv_w    = (const float*)d_in[3];
    const float* qkv_b    = (const float*)d_in[4];
    const float* proj_w   = (const float*)d_in[5];
    const float* proj_b   = (const float*)d_in[6];
    const float* rel_tab  = (const float*)d_in[7];
    const float* norm2_w  = (const float*)d_in[8];
    const float* norm2_b  = (const float*)d_in[9];
    const float* fc1_w    = (const float*)d_in[10];
    const float* fc1_b    = (const float*)d_in[11];
    const float* fc2_w    = (const float*)d_in[12];
    const float* fc2_b    = (const float*)d_in[13];
    const int*   rel_idx  = (const int*)d_in[14];
    float* out = (float*)d_out;

    // ---- workspace carve (all chunks 256B-aligned multiples) ----
    char* p = (char*)d_ws;
    __bf16* wq  = (__bf16*)p; p += (long)3 * CH * CH * 2;           // 1.5 MB
    __bf16* wp  = (__bf16*)p; p += (long)CH * CH * 2;               // 0.5 MB
    __bf16* w1  = (__bf16*)p; p += (long)HID * CH * 2;              // 2 MB
    __bf16* w2  = (__bf16*)p; p += (long)CH * HID * 2;              // 2 MB
    float*  bpd = (float*)p;  p += (long)HEADS * 64 * 64 * 4;       // 256 KB
    __bf16* xw  = (__bf16*)p; p += (long)MWIN * CH * 2;             // 64 MB
    __bf16* qkv = (__bf16*)p; p += (long)MWIN * 3 * CH * 2;         // 192 MB
    __bf16* ato = (__bf16*)p; p += (long)MWIN * CH * 2;             // 64 MB
    float*  prj = (float*)p;  p += (long)MWIN * CH * 4;             // 128 MB
    float*  xat = (float*)p;  p += (long)MTOK * CH * 4;             // 100 MB
    __bf16* l2b = (__bf16*)p; p += (long)MTOK * CH * 2;             // 50 MB
    __bf16* f1o = (__bf16*)p; p += (long)MTOK * HID * 2;            // 200 MB

    // 1) weight conversion f32 -> bf16, padded bias build
    cvt_bf16_kernel<<<3072, 256, 0, stream>>>(qkv_w, wq, (long)3 * CH * CH);
    cvt_bf16_kernel<<<1024, 256, 0, stream>>>(proj_w, wp, (long)CH * CH);
    cvt_bf16_kernel<<<4096, 256, 0, stream>>>(fc1_w, w1, (long)HID * CH);
    cvt_bf16_kernel<<<4096, 256, 0, stream>>>(fc2_w, w2, (long)CH * HID);
    bias_pad_kernel<<<HEADS * 4096 / 256, 256, 0, stream>>>(rel_tab, rel_idx, bpd);

    // 2) LN1 + window partition (pad 49->64)
    ln1_window_kernel<<<MWIN / 8, 256, 0, stream>>>(x, norm1_w, norm1_b, xw);

    // 3) QKV: (65536 x 512) @ (1536 x 512)^T
    gemm_bf16_kernel<false, true, false>
        <<<dim3(3 * CH / 32, MWIN / 128), 256, 0, stream>>>(
            xw, wq, qkv_b, qkv, nullptr, MWIN, 3 * CH, CH);

    // 4) attention per (window, head)
    attn_kernel<<<NWIN * HEADS, 128, 0, stream>>>(qkv, bpd, ato);

    // 5) proj: (65536 x 512) @ (512 x 512)^T -> f32
    gemm_bf16_kernel<false, false, false>
        <<<dim3(CH / 32, MWIN / 128), 256, 0, stream>>>(
            ato, wp, proj_b, prj, nullptr, MWIN, CH, CH);

    // 6) window reverse + residual
    revres_kernel<<<MTOK / 8, 256, 0, stream>>>(x, prj, xat);

    // 7) LN2
    ln2_kernel<<<MTOK / 8, 256, 0, stream>>>(xat, norm2_w, norm2_b, l2b);

    // 8) FC1 + GELU: (50176 x 512) @ (2048 x 512)^T -> bf16
    gemm_bf16_kernel<true, true, false>
        <<<dim3(HID / 32, MTOK / 128), 256, 0, stream>>>(
            l2b, w1, fc1_b, f1o, nullptr, MTOK, HID, CH);

    // 9) FC2 + residual: (50176 x 2048) @ (512 x 2048)^T + xat -> d_out (f32)
    gemm_bf16_kernel<false, false, true>
        <<<dim3(CH / 32, MTOK / 128), 256, 0, stream>>>(
            f1o, w2, fc2_b, out, xat, MTOK, CH, HID);
}
